// BasicNCAModel_83348135346373
// MI455X (gfx1250) — matrix operand order repaired and checked by reading, not compiled, not run on hardware
//
#include <hip/hip_runtime.h>

typedef __attribute__((ext_vector_type(16))) _Float16 v16h;
typedef __attribute__((ext_vector_type(8)))  _Float16 v8h;
typedef __attribute__((ext_vector_type(2)))  _Float16 v2h;
typedef __attribute__((ext_vector_type(8)))  float    v8f;

#define HH 256
#define WW 256
#define CC 32
#define NB 16
#define TW 16
#define TH 8
#define XS_W 18            // TW+2 (halo)
#define XS_H 10            // TH+2
#define XS_STR 33          // 32 ch + 1 pad float -> conflict-free
#define Y_STR 104          // 96 feat + 8 pad f16 (keeps 16B row alignment)
#define W1_STR 104         // 96 K + pad
#define W2_STR 264         // 256 K + pad
#define H1_STR 40          // 32 + pad, 80B rows (16B aligned)

union frag16 { v16h v; v8h h[2]; };

__global__ __launch_bounds__(256)
void nca_step_kernel(const float* __restrict__ xin,
                     float* __restrict__ xout,
                     const float* __restrict__ f1g,
                     const float* __restrict__ f2g,
                     const float* __restrict__ w1g,
                     const float* __restrict__ b1g,
                     const float* __restrict__ w2g,
                     const float* __restrict__ stoch,
                     int step)
{
    __shared__ __align__(16) float    xs [XS_H*XS_W*XS_STR]; // x tile + halo (f32)
    __shared__ __align__(16) _Float16 ysm[TH*TW*Y_STR];      // perception feats (f16)
    __shared__ __align__(16) _Float16 w1t[256*W1_STR];       // W1^T  [N=256][K=96]
    __shared__ __align__(16) _Float16 w2t[32*W2_STR];        // W2^T  [N=32][K=256]
    __shared__ __align__(16) _Float16 h1s[8*16*H1_STR];      // per-wave h1 staging
    __shared__ float b1s[256];
    __shared__ float f1s[9*32];
    __shared__ float f2s[9*32];
    __shared__ float maskv[TH*TW];

    const int t  = threadIdx.x;
    const int tx = blockIdx.x, ty = blockIdx.y, bb = blockIdx.z;

    // ---------------- Phase 1a: stage x tile (reflect halo) via CDNA5 async-to-LDS.
    // Per-lane global gather address + per-lane padded LDS offset; tracked on ASYNCcnt.
    for (int idx = t; idx < XS_H*XS_W*CC; idx += 256) {
        int c = idx & 31, pos = idx >> 5;
        int lx = pos % XS_W, ly = pos / XS_W;
        int gx = tx*TW + lx - 1, gy = ty*TH + ly - 1;
        gx = gx < 0 ? -gx : (gx > WW-1 ? 2*(WW-1)-gx : gx);  // reflect pad
        gy = gy < 0 ? -gy : (gy > HH-1 ? 2*(HH-1)-gy : gy);
        const float* gp = &xin[(((long)bb*HH + gy)*WW + gx)*CC + c];
        unsigned lo = (unsigned)(uintptr_t)(&xs[pos*XS_STR + c]); // flat[31:0] == LDS offset
        asm volatile("global_load_async_to_lds_b32 %0, %1, off"
                     :: "v"(lo), "v"(gp) : "memory");
    }
    // Weights need f32->f16 conversion + transpose => normal path.
    for (int idx = t; idx < 96*256; idx += 256) {             // W1[K][N] -> W1^T f16
        int K = idx >> 8, N = idx & 255;
        w1t[N*W1_STR + K] = (_Float16)w1g[idx];
    }
    for (int idx = t; idx < 256*32; idx += 256) {             // W2[K][N] -> W2^T f16
        int K = idx >> 5, N = idx & 31;
        w2t[N*W2_STR + K] = (_Float16)w2g[idx];
    }
    b1s[t] = b1g[t];
    if (t < 288) { f1s[t] = f1g[t]; f2s[t] = f2g[t]; }
    if (t < TH*TW) {
        int px = t & 15, py = t >> 4;
        int gx = tx*TW + px, gy = ty*TH + py;
        float sv = stoch[(((long)step*NB + bb)*HH + gy)*WW + gx];
        maskv[t] = sv > 0.5f ? 1.0f : 0.0f;
    }
    asm volatile("s_wait_asynccnt 0" ::: "memory");           // async tile copy done
    __syncthreads();

    // ---------------- Phase 1b: perception -> ysm[pix][96] = [x | dw1 | dw2] (f16)
    {
        int p  = t >> 1, h = t & 1;                 // 2 threads per pixel, 16 ch each
        int px = p & 15, py = p >> 4;
        int cpos = (py+1)*XS_W + (px+1);
        int cb = h*16;
        for (int cc = 0; cc < 16; cc += 2) {
            int c = cb + cc;
            float i0 = xs[cpos*XS_STR + c], i1 = xs[cpos*XS_STR + c + 1];
            float a0 = 0.f, a1 = 0.f, c0 = 0.f, c1 = 0.f;
            #pragma unroll
            for (int tap = 0; tap < 9; ++tap) {
                int np = cpos + (tap/3 - 1)*XS_W + (tap%3 - 1);
                float x0 = xs[np*XS_STR + c], x1 = xs[np*XS_STR + c + 1];
                a0 += f1s[tap*32 + c]     * x0;
                a1 += f1s[tap*32 + c + 1] * x1;
                c0 += f2s[tap*32 + c]     * x0;
                c1 += f2s[tap*32 + c + 1] * x1;
            }
            v2h pk;
            pk[0] = (_Float16)i0; pk[1] = (_Float16)i1;
            *(v2h*)&ysm[p*Y_STR + c]       = pk;
            pk[0] = (_Float16)a0; pk[1] = (_Float16)a1;
            *(v2h*)&ysm[p*Y_STR + 32 + c]  = pk;
            pk[0] = (_Float16)c0; pk[1] = (_Float16)c1;
            *(v2h*)&ysm[p*Y_STR + 64 + c]  = pk;
        }
    }
    __syncthreads();

    // ---------------- Phase 2: per-wave fused MLP via WMMA (M-tile = 16 pixels)
    const int wave = t >> 5, lane = t & 31;
    const int hi = lane >> 4, ln = lane & 15;
    const int r = wave;                               // tile row = this wave's M-tile

    // A fragments for GEMM1: 16x96 f16, 3 K-chunks of 32
    frag16 a1[3];
    #pragma unroll
    for (int kc = 0; kc < 3; ++kc) {
        const _Float16* yp = &ysm[(r*16 + ln)*Y_STR + kc*32 + 8*hi];
        a1[kc].h[0] = *(const v8h*)yp;                // K = kc*32 + 8*hi + [0..7]
        a1[kc].h[1] = *(const v8h*)(yp + 16);         // K = kc*32 + 16 + 8*hi + [0..7]
    }

    v8f acc2[2]; acc2[0] = {}; acc2[1] = {};
    _Float16* stg = &h1s[wave*16*H1_STR];             // private to this wave

    for (int j2 = 0; j2 < 8; ++j2) {                  // GEMM2 K-chunk = 2 GEMM1 N-tiles
        #pragma unroll
        for (int jj = 0; jj < 2; ++jj) {
            int j = j2*2 + jj;                        // GEMM1 N-tile (0..15)
            // bias folded into the WMMA C operand: D = A*B + b1 (per-lane N bias)
            float bias = b1s[j*16 + ln];
            v8f acc;
            #pragma unroll
            for (int i = 0; i < 8; ++i) acc[i] = bias;
            #pragma unroll
            for (int kc = 0; kc < 3; ++kc) {
                frag16 bf;
                const _Float16* wp = &w1t[(j*16 + ln)*W1_STR + kc*32 + 8*hi];
                bf.h[0] = *(const v8h*)wp;            // K = kc*32 + 8*hi + [0..7]
                bf.h[1] = *(const v8h*)(wp + 16);     // K = kc*32 + 16 + 8*hi + [0..7]
                acc = __builtin_amdgcn_wmma_f32_16x16x32_f16(
                        false, a1[kc].v, false, bf.v, (short)0, acc, false, false);
            }
            #pragma unroll
            for (int v = 0; v < 8; ++v) {             // D: lane=N, vgpr=M -> relu, f16
                float f = acc[v];
                f = f > 0.f ? f : 0.f;
                stg[(v + 8*hi)*H1_STR + jj*16 + ln] = (_Float16)f;
            }
        }
        // reload as GEMM2 A fragment (lane = M); same-wave DS ops are in-order
        frag16 a2;
        const _Float16* sp = &stg[ln*H1_STR + 8*hi];
        a2.h[0] = *(const v8h*)sp;
        a2.h[1] = *(const v8h*)(sp + 16);
        #pragma unroll
        for (int n2 = 0; n2 < 2; ++n2) {
            frag16 b2;
            const _Float16* wp = &w2t[(n2*16 + ln)*W2_STR + j2*32 + 8*hi];
            b2.h[0] = *(const v8h*)wp;
            b2.h[1] = *(const v8h*)(wp + 16);
            acc2[n2] = __builtin_amdgcn_wmma_f32_16x16x32_f16(
                         false, a2.v, false, b2.v, (short)0, acc2[n2], false, false);
        }
    }

    // ---------------- Epilogue: gating + residual + store
    #pragma unroll
    for (int v = 0; v < 8; ++v) {
        int pxv = v + 8*hi;                            // pixel (M) for this VGPR
        float mk = maskv[r*16 + pxv];
        int gx = tx*TW + pxv, gy = ty*TH + r;
        long obase = (((long)bb*HH + gy)*WW + gx)*CC;
        int xpos = ((r+1)*XS_W + (pxv+1))*XS_STR;
        #pragma unroll
        for (int t2 = 0; t2 < 2; ++t2) {
            int ch = t2*16 + ln;                       // channel (N) on lanes
            float dx = acc2[t2][v] * mk;
            if (t2 == 0 && ln < 3) dx = 0.f;           // immutable image channels
            xout[obase + ch] = xs[xpos + ch] + dx;
        }
    }
}

extern "C" void kernel_launch(void* const* d_in, const int* in_sizes, int n_in,
                              void* d_out, int out_size, void* d_ws, size_t ws_size,
                              hipStream_t stream) {
    (void)in_sizes; (void)n_in; (void)out_size; (void)ws_size;
    const float* x  = (const float*)d_in[0];
    const float* f1 = (const float*)d_in[1];
    const float* f2 = (const float*)d_in[2];
    const float* w1 = (const float*)d_in[3];
    const float* b1 = (const float*)d_in[4];
    const float* w2 = (const float*)d_in[5];
    const float* st = (const float*)d_in[6];
    float* out = (float*)d_out;
    float* tmp = (float*)d_ws;   // one x-sized (128 MB) ping-pong buffer

    dim3 grid(WW/TW, HH/TH, NB);
    dim3 block(256);
    // 4 NCA steps with global sync between (spatial conv needs halo exchange):
    // x -> tmp -> out -> tmp -> out
    nca_step_kernel<<<grid, block, 0, stream>>>(x,   tmp, f1, f2, w1, b1, w2, st, 0);
    nca_step_kernel<<<grid, block, 0, stream>>>(tmp, out, f1, f2, w1, b1, w2, st, 1);
    nca_step_kernel<<<grid, block, 0, stream>>>(out, tmp, f1, f2, w1, b1, w2, st, 2);
    nca_step_kernel<<<grid, block, 0, stream>>>(tmp, out, f1, f2, w1, b1, w2, st, 3);
}
